// BlockRC3_16140487099015
// MI455X (gfx1250) — matrix operand order repaired and checked by reading, not compiled, not run on hardware
//
#include <hip/hip_runtime.h>
#include <hip/hip_bf16.h>
#include <math.h>

// ---------------------------------------------------------------------------
// MI455X (gfx1250): all contractions run through v_wmma_f32_16x16x32_f16.
// f16 row-major A tiles are staged into LDS with global_load_async_to_lds_b128
// (ASYNCcnt-tracked, no VGPR round trip); B tiles are staged transposed so
// both A and B fragments load as contiguous ds_load_b128 pairs.
// ---------------------------------------------------------------------------

typedef __attribute__((ext_vector_type(16))) _Float16 v16h;
typedef __attribute__((ext_vector_type(8)))  float    v8f;

#define EMB 320
#define MDIM 160
#define TPB 16384                 // tokens per batch (128x128)
#define TT  65536                 // total tokens (B=4)

__device__ __forceinline__ float gelu_exact(float v) {
    return 0.5f * v * (1.0f + erff(v * 0.70710678118654752f));
}

// CDNA5 async copy: 16 bytes global -> LDS per active lane (ASYNCcnt).
__device__ __forceinline__ void async_ld_b128(void* lds_ptr, const void* gptr) {
    unsigned l = (unsigned)(unsigned long long)lds_ptr;   // low 32 bits = LDS byte addr
    asm volatile("global_load_async_to_lds_b128 %0, %1, off"
                 :: "v"(l), "v"(gptr)
                 : "memory");
}
__device__ __forceinline__ void wait_async() {
    asm volatile("s_wait_asynccnt 0" ::: "memory");
}

// ---- WMMA fragment loads from LDS (wave32 layouts, ISA 7.12.2) -------------
// A 16x32 f16: lanes 0-15 row M=lane, VGPR0-3 K=0..7, VGPR4-7 K=16..23;
//              lanes 16-31 same rows, K+=8. Two contiguous 16B runs per lane.
__device__ __forceinline__ v16h frag_a(const _Float16* tile, int stride, int lane) {
    const _Float16* p = tile + (lane & 15) * stride + ((lane & 16) >> 1);
    v16h a;
#pragma unroll
    for (int j = 0; j < 16; ++j)
        a[j] = p[((j & 8) << 1) + (j & 7)];
    return a;
}
// B 32x16 f16 from transposed LDS tile BsT[n][k]: lanes 0-15 K=0..7,16..23,
// lanes 16-31 K+=8 -> two contiguous 16B runs per lane (same order as A).
__device__ __forceinline__ v16h frag_b_t(const _Float16* BsT, int ncol, int stride, int lane) {
    const _Float16* p = BsT + (ncol + (lane & 15)) * stride + ((lane & 16) >> 1);
    v16h b;
#pragma unroll
    for (int j = 0; j < 16; ++j)
        b[j] = p[((j & 8) << 1) + (j & 7)];
    return b;
}

// ---- Generic tiled WMMA GEMM ----------------------------------------------
// Block tile 128x64x32, 256 threads (8 waves), wave tile 32x32 (2x2 frags).
// ASYNC_A: A source is f16 row-major & M multiple of 128 -> async block copy.
template <bool ASYNC_A, class AL, class BL, class EP>
__global__ __launch_bounds__(256) void gemm_wmma(AL aload, BL bload, EP ep,
                                                 int Mdim, int Ndim, int Kdim) {
    constexpr int BM = 128, BN = 64, BK = 32;
    constexpr int AST = BK + 16;   // 48 halfs = 96B rows: every 16B seg aligned
    constexpr int BST = BK + 16;
    __shared__ __align__(16) _Float16 As[BM * AST];
    __shared__ __align__(16) _Float16 BsT[BN * BST];

    const int tid  = threadIdx.x;
    const int lane = tid & 31;
    const int wave = tid >> 5;
    const int bz   = blockIdx.z;
    const int bm0  = blockIdx.x * BM;
    const int bn0  = blockIdx.y * BN;
    const int wm   = (wave & 3) * 32;
    const int wn   = (wave >> 2) * 32;

    v8f acc[2][2];
    const v8f vzero = {};
#pragma unroll
    for (int mi = 0; mi < 2; ++mi)
#pragma unroll
        for (int ni = 0; ni < 2; ++ni) acc[mi][ni] = vzero;

    for (int k0 = 0; k0 < Kdim; k0 += BK) {
        // ---- stage A tile ----
        if constexpr (ASYNC_A) {
            // 128 rows x 64B: 512 x b128 async copies, 2 per thread
#pragma unroll
            for (int i = tid; i < BM * 4; i += 256) {
                int m   = i >> 2;
                int seg = (i & 3) << 3;                       // half offset 0,8,16,24
                const _Float16* g = aload.rowptr(bz, (long)(bm0 + m)) + (k0 + seg);
                async_ld_b128(&As[m * AST + seg], g);
            }
        } else {
            for (int i = tid; i < BM * BK; i += 256) {
                int m = i >> 5, k = i & 31;
                As[m * AST + k] = (_Float16)aload(bz, (long)(bm0 + m), (long)(k0 + k));
            }
        }
        // ---- stage B tile, transposed: BsT[n][k] ----
        for (int i = tid; i < BK * BN; i += 256) {
            int n = i >> 5, k = i & 31;
            BsT[n * BST + k] = (_Float16)bload(bz, (long)(k0 + k), (long)(bn0 + n));
        }
        if constexpr (ASYNC_A) wait_async();
        __syncthreads();

        v16h bf0 = frag_b_t(BsT, wn,      BST, lane);
        v16h bf1 = frag_b_t(BsT, wn + 16, BST, lane);
#pragma unroll
        for (int mi = 0; mi < 2; ++mi) {
            v16h af = frag_a(&As[(wm + mi * 16) * AST], AST, lane);
            acc[mi][0] = __builtin_amdgcn_wmma_f32_16x16x32_f16(
                false, af, false, bf0, (short)0, acc[mi][0], false, false);
            acc[mi][1] = __builtin_amdgcn_wmma_f32_16x16x32_f16(
                false, af, false, bf1, (short)0, acc[mi][1], false, false);
        }
        __syncthreads();
    }

    // C frag: lane&15 = column, VGPR r = row r + 8*(lane>=16)
    const int mhi = (lane >> 4) << 3;
#pragma unroll
    for (int mi = 0; mi < 2; ++mi)
#pragma unroll
        for (int ni = 0; ni < 2; ++ni)
#pragma unroll
            for (int r = 0; r < 8; ++r) {
                int m = bm0 + wm + mi * 16 + r + mhi;
                int n = bn0 + wn + ni * 16 + (lane & 15);
                if (m < Mdim && n < Ndim) ep(bz, (long)m, (long)n, acc[mi][ni][r]);
            }
}

// ---- A/B loader functors ---------------------------------------------------
struct ALdConv {                 // implicit im2col: row=(b,oy,ox), k=(ky,kx,ci)
    const float* x; int dil, pad;
    __device__ float operator()(int, long m, long k) const {
        int b  = (int)(m >> 14);
        int p  = (int)(m & 16383);
        int oy = p >> 7, ox = p & 127;
        int kk = (int)k;
        int ky  = kk / 192;
        int rem = kk - ky * 192;
        int kx  = rem >> 6;
        int ci  = rem & 63;
        int iy = oy * 2 + ky * dil - pad;
        int ix = ox * 2 + kx * dil - pad;
        if ((unsigned)iy >= 256u || (unsigned)ix >= 256u) return 0.0f;
        return x[(((((long)b << 16) | ((long)iy << 8) | (long)ix)) << 6) | ci];
    }
};
struct BLdConvW {                // w[o][ci][ky][kx], n=o, k=(ky,kx,ci)
    const float* w;
    __device__ float operator()(int, long k, long n) const {
        if (n >= 160) return 0.0f;
        int kk = (int)k;
        int ky  = kk / 192;
        int rem = kk - ky * 192;
        int kx  = rem >> 6;
        int ci  = rem & 63;
        return w[((n * 64 + ci) * 3 + ky) * 3 + kx];
    }
};
struct ALdF16 {                  // row-major f16 A(m,k) (async-capable)
    const _Float16* a; long ld, bstride, rows;
    __device__ const _Float16* rowptr(int b, long m) const {
        return a + (long)b * bstride + m * ld;
    }
    __device__ float operator()(int b, long m, long k) const {
        if (m >= rows) return 0.0f;
        return (float)a[(long)b * bstride + m * ld + k];
    }
};
struct ALdF16T {                 // A(m,k) = a[k*ld + m]  (transposed source)
    const _Float16* a; long ld, bstride, cols;
    __device__ const _Float16* rowptr(int, long) const { return a; }  // unused
    __device__ float operator()(int b, long m, long k) const {
        if (m >= cols) return 0.0f;
        return (float)a[(long)b * bstride + k * ld + m];
    }
};
struct BLdF32T {                 // B(k,n) = w[n*ld + k]  (fp32 weights, W^T)
    const float* w; long ld, nmax, bstride;
    __device__ float operator()(int b, long k, long n) const {
        if (n >= nmax) return 0.0f;
        const float* p = &w[(long)b * bstride + n * ld + k];
        __builtin_prefetch(p + 32, 0, 1);   // next K tile (global_prefetch)
        return *p;
    }
};
struct BLdF16 {                  // row-major f16 B(k,n)
    const _Float16* p; long ld, nmax, bstride;
    __device__ float operator()(int b, long k, long n) const {
        if (n >= nmax) return 0.0f;
        return (float)p[(long)b * bstride + k * ld + n];
    }
};

// ---- Epilogue functors -----------------------------------------------------
struct EpConvGelu {              // t[m, choff+n] = gelu(acc + bias[n])
    float* t; const float* bias; int choff;
    __device__ void operator()(int, long m, long n, float v) const {
        t[m * EMB + choff + n] = gelu_exact(v + bias[n]);
    }
};
struct EpKQV {                   // split 960 -> k16 | q16 | v(32+16)
    _Float16* k16; _Float16* q16; float* v32; _Float16* v16; const float* bias;
    __device__ void operator()(int, long m, long n, float v) const {
        v += bias[n];
        if (n < EMB)            k16[m * EMB + n]         = (_Float16)v;
        else if (n < 2 * EMB)   q16[m * EMB + (n - EMB)] = (_Float16)v;
        else { long i = m * EMB + (n - 2 * EMB); v32[i] = v; v16[i] = (_Float16)v; }
    }
};
struct EpPrm {                   // exp(acc - xd[m]) / sqrt(M)
    _Float16* out; const float* xd;
    __device__ void operator()(int, long m, long n, float v) const {
        out[m * MDIM + n] = (_Float16)(expf(v - xd[m]) * 0.07905694150420949f);
    }
};
struct EpKptv {                  // kptv[b][e=m][mcol=n]
    float* o;
    __device__ void operator()(int b, long m, long n, float v) const {
        o[((long)b * EMB + m) * MDIM + n] = v;
    }
};
struct EpYa {                    // acc / (D[row]+eps) -> f16
    _Float16* out; const float* D; long rpb;
    __device__ void operator()(int b, long m, long n, float v) const {
        long r = (long)b * rpb + m;
        out[r * EMB + n] = (_Float16)(v / (D[r] + 1e-8f));
    }
};
struct EpProj {                  // ya = v + acc + bias
    float* ya; const float* v32; const float* bias;
    __device__ void operator()(int, long m, long n, float v) const {
        long i = m * EMB + n;
        ya[i] = v32[i] + v + bias[n];
    }
};
struct EpGelu16 {                // gelu(acc + bias) -> f16
    _Float16* o; const float* bias;
    __device__ void operator()(int, long m, long n, float v) const {
        o[m * EMB + n] = (_Float16)gelu_exact(v + bias[n]);
    }
};
struct EpOut {                   // out = ya + acc + bias
    float* o; const float* ya; const float* bias;
    __device__ void operator()(int, long m, long n, float v) const {
        long i = m * EMB + n;
        o[i] = ya[i] + v + bias[n];
    }
};

// ---- Small wave-per-row helpers -------------------------------------------
__global__ __launch_bounds__(256) void layernorm_f16(const float* __restrict__ in,
                                                     const float* __restrict__ g,
                                                     const float* __restrict__ b,
                                                     _Float16* __restrict__ out,
                                                     long rows) {
    int wave = threadIdx.x >> 5, lane = threadIdx.x & 31;
    long row = (long)blockIdx.x * 8 + wave;
    if (row >= rows) return;
    const float* p = in + row * EMB;
    float s = 0.f, s2 = 0.f;
    for (int i = lane; i < EMB; i += 32) { float v = p[i]; s += v; s2 += v * v; }
#pragma unroll
    for (int off = 16; off; off >>= 1) {
        s  += __shfl_xor(s,  off, 32);
        s2 += __shfl_xor(s2, off, 32);
    }
    float mu   = s * (1.0f / EMB);
    float var  = s2 * (1.0f / EMB) - mu * mu;
    float rstd = rsqrtf(var + 1e-5f);
    _Float16* q = out + row * EMB;
    for (int i = lane; i < EMB; i += 32)
        q[i] = (_Float16)((p[i] - mu) * rstd * g[i] + b[i]);
}

__global__ __launch_bounds__(256) void rowsumsq_f16(const _Float16* __restrict__ z,
                                                    float* __restrict__ xd, long rows) {
    int wave = threadIdx.x >> 5, lane = threadIdx.x & 31;
    long row = (long)blockIdx.x * 8 + wave;
    if (row >= rows) return;
    const _Float16* p = z + row * EMB;
    float s = 0.f;
    for (int i = lane; i < EMB; i += 32) { float v = (float)p[i]; s += v * v; }
#pragma unroll
    for (int off = 16; off; off >>= 1) s += __shfl_xor(s, off, 32);
    if (lane == 0) xd[row] = 0.5f * s;
}

__global__ __launch_bounds__(256) void ksum_kernel(const _Float16* __restrict__ kp,
                                                   float* __restrict__ ksum) {
    int m = blockIdx.x, b = blockIdx.y;
    const _Float16* p = kp + ((long)b * TPB) * MDIM + m;
    float s = 0.f;
    for (int t = threadIdx.x; t < TPB; t += 256) s += (float)p[(long)t * MDIM];
    __shared__ float sm[256];
    sm[threadIdx.x] = s;
    __syncthreads();
    for (int off = 128; off; off >>= 1) {
        if (threadIdx.x < off) sm[threadIdx.x] += sm[threadIdx.x + off];
        __syncthreads();
    }
    if (threadIdx.x == 0) ksum[b * MDIM + m] = sm[0];
}

__global__ __launch_bounds__(256) void dvec_kernel(const _Float16* __restrict__ qp,
                                                   const float* __restrict__ ksum,
                                                   float* __restrict__ D, long rows) {
    int wave = threadIdx.x >> 5, lane = threadIdx.x & 31;
    long row = (long)blockIdx.x * 8 + wave;
    if (row >= rows) return;
    int b = (int)(row >> 14);
    const _Float16* p = qp + row * MDIM;
    const float* ks = ksum + (long)b * MDIM;
    float s = 0.f;
    for (int i = lane; i < MDIM; i += 32) s += (float)p[i] * ks[i];
#pragma unroll
    for (int off = 16; off; off >>= 1) s += __shfl_xor(s, off, 32);
    if (lane == 0) D[row] = s;
}

// ---------------------------------------------------------------------------
extern "C" void kernel_launch(void* const* d_in, const int* /*in_sizes*/, int /*n_in*/,
                              void* d_out, int /*out_size*/, void* d_ws, size_t /*ws_size*/,
                              hipStream_t stream) {
    const float* x     = (const float*)d_in[0];
    const float* cw1   = (const float*)d_in[1];
    const float* cb1   = (const float*)d_in[2];
    const float* cw2   = (const float*)d_in[3];
    const float* cb2   = (const float*)d_in[4];
    const float* ln1g  = (const float*)d_in[5];
    const float* ln1b  = (const float*)d_in[6];
    const float* kqvw  = (const float*)d_in[7];
    const float* kqvb  = (const float*)d_in[8];
    const float* projw = (const float*)d_in[9];
    const float* projb = (const float*)d_in[10];
    const float* perfw = (const float*)d_in[11];
    const float* ln2g  = (const float*)d_in[12];
    const float* ln2b  = (const float*)d_in[13];
    const float* mw1   = (const float*)d_in[14];
    const float* mb1   = (const float*)d_in[15];
    const float* mw2   = (const float*)d_in[16];
    const float* mb2   = (const float*)d_in[17];

    char* w = (char*)d_ws;
    auto carve = [&](size_t bytes) {
        char* p = w;
        w += (bytes + 255) & ~(size_t)255;
        return p;
    };
    float*     t32    = (float*)    carve((size_t)TT * EMB * 4);
    _Float16*  h16    = (_Float16*) carve((size_t)TT * EMB * 2);
    _Float16*  k16    = (_Float16*) carve((size_t)TT * EMB * 2);
    _Float16*  q16    = (_Float16*) carve((size_t)TT * EMB * 2);
    float*     v32    = (float*)    carve((size_t)TT * EMB * 4);
    _Float16*  v16    = (_Float16*) carve((size_t)TT * EMB * 2);
    float*     xdk    = (float*)    carve((size_t)TT * 4);
    float*     xdq    = (float*)    carve((size_t)TT * 4);
    _Float16*  kp16   = (_Float16*) carve((size_t)TT * MDIM * 2);
    _Float16*  qp16   = (_Float16*) carve((size_t)TT * MDIM * 2);
    float*     ksum   = (float*)    carve((size_t)4 * MDIM * 4);
    float*     kptv32 = (float*)    carve((size_t)4 * EMB * MDIM * 4);
    float*     Dbuf   = (float*)    carve((size_t)TT * 4);
    _Float16*  yap16  = (_Float16*) carve((size_t)TT * EMB * 2);
    float*     ya32   = (float*)    carve((size_t)TT * EMB * 4);
    _Float16*  h2_16  = (_Float16*) carve((size_t)TT * EMB * 2);
    _Float16*  u16    = (_Float16*) carve((size_t)TT * EMB * 2);

    const dim3 blk(256);

    // 1) convs as implicit-im2col WMMA GEMMs: M=65536, N=160, K=576
    {
        ALdConv a1{x, 1, 1}; BLdConvW b1{cw1}; EpConvGelu e1{t32, cb1, 0};
        gemm_wmma<false><<<dim3(TT / 128, 3, 1), blk, 0, stream>>>(a1, b1, e1, TT, 160, 576);
        ALdConv a2{x, 2, 2}; BLdConvW b2{cw2}; EpConvGelu e2{t32, cb2, 160};
        gemm_wmma<false><<<dim3(TT / 128, 3, 1), blk, 0, stream>>>(a2, b2, e2, TT, 160, 576);
    }
    // 2) LN1 -> h16
    layernorm_f16<<<dim3(TT / 8), blk, 0, stream>>>(t32, ln1g, ln1b, h16, TT);
    // 3) kqv GEMM: 65536 x 960 x 320 (async A staging)
    {
        ALdF16 a{h16, EMB, 0, TT};
        BLdF32T b{kqvw, EMB, 3 * EMB, 0};
        EpKQV e{k16, q16, v32, v16, kqvb};
        gemm_wmma<true><<<dim3(TT / 128, (3 * EMB) / 64, 1), blk, 0, stream>>>(a, b, e, TT, 3 * EMB, EMB);
    }
    // 4) xd = 0.5*|z|^2 for k and q
    rowsumsq_f16<<<dim3(TT / 8), blk, 0, stream>>>(k16, xdk, TT);
    rowsumsq_f16<<<dim3(TT / 8), blk, 0, stream>>>(q16, xdq, TT);
    // 5) performer features: 65536 x 160 x 320, exp epilogue (async A)
    {
        ALdF16 ak{k16, EMB, 0, TT};
        BLdF32T bp{perfw, EMB, MDIM, 0};
        EpPrm ek{kp16, xdk};
        gemm_wmma<true><<<dim3(TT / 128, 3, 1), blk, 0, stream>>>(ak, bp, ek, TT, MDIM, EMB);
        ALdF16 aq{q16, EMB, 0, TT};
        EpPrm eq{qp16, xdq};
        gemm_wmma<true><<<dim3(TT / 128, 3, 1), blk, 0, stream>>>(aq, bp, eq, TT, MDIM, EMB);
    }
    // 6) ksum[b][m] = sum_t kp
    ksum_kernel<<<dim3(MDIM, 4), blk, 0, stream>>>(kp16, ksum);
    // 7) kptv[b] = v^T kp : per batch 320 x 160 x 16384 (transposed A: scalar)
    {
        ALdF16T a{v16, EMB, (long)TPB * EMB, EMB};
        BLdF16  b{kp16, MDIM, MDIM, (long)TPB * MDIM};
        EpKptv  e{kptv32};
        gemm_wmma<false><<<dim3(3, 3, 4), blk, 0, stream>>>(a, b, e, EMB, MDIM, TPB);
    }
    // 8) D[t] = qp . ksum[b]
    dvec_kernel<<<dim3(TT / 8), blk, 0, stream>>>(qp16, ksum, Dbuf, TT);
    // 9) ya_pre = (qp @ kptv^T) / (D+eps) : per batch 16384 x 320 x 160 (async A)
    {
        ALdF16  a{qp16, MDIM, (long)TPB * MDIM, TPB};
        BLdF32T b{kptv32, MDIM, EMB, (long)EMB * MDIM};
        EpYa    e{yap16, Dbuf, TPB};
        gemm_wmma<true><<<dim3(TPB / 128, EMB / 64, 4), blk, 0, stream>>>(a, b, e, TPB, EMB, MDIM);
    }
    // 10) proj + residual: ya = v + ya_pre @ proj_w^T + b (async A)
    {
        ALdF16  a{yap16, EMB, 0, TT};
        BLdF32T b{projw, EMB, EMB, 0};
        EpProj  e{ya32, v32, projb};
        gemm_wmma<true><<<dim3(TT / 128, EMB / 64, 1), blk, 0, stream>>>(a, b, e, TT, EMB, EMB);
    }
    // 11) LN2
    layernorm_f16<<<dim3(TT / 8), blk, 0, stream>>>(ya32, ln2g, ln2b, h2_16, TT);
    // 12) MLP fc1 + gelu (async A)
    {
        ALdF16   a{h2_16, EMB, 0, TT};
        BLdF32T  b{mw1, EMB, EMB, 0};
        EpGelu16 e{u16, mb1};
        gemm_wmma<true><<<dim3(TT / 128, EMB / 64, 1), blk, 0, stream>>>(a, b, e, TT, EMB, EMB);
    }
    // 13) MLP fc2 + residual -> d_out (async A)
    {
        ALdF16  a{u16, EMB, 0, TT};
        BLdF32T b{mw2, EMB, EMB, 0};
        EpOut   e{(float*)d_out, ya32, mb2};
        gemm_wmma<true><<<dim3(TT / 128, EMB / 64, 1), blk, 0, stream>>>(a, b, e, TT, EMB, EMB);
    }
}
